// LinearSpatialGVA_70574902608065
// MI455X (gfx1250) — matrix operand order repaired and checked by reading, not compiled, not run on hardware
//
#include <hip/hip_runtime.h>

// ---------------------------------------------------------------------------
// Linear spatial attention, MI455X (gfx1250) — bf16 WMMA path, f32 accumulate.
//   B=8, N=65536, C=D=128
// ---------------------------------------------------------------------------

#define DEVFN __device__ __forceinline__

typedef __attribute__((ext_vector_type(16))) __bf16 v16bf;
typedef __attribute__((ext_vector_type(8)))  float  v8f;

constexpr int Bn = 8;
constexpr int Nn = 65536;
constexpr int Cn = 128;
constexpr int SLABS = 4;          // 128-row slabs per block in kernel 1

DEVFN v8f wmma_bf16(v16bf a, v16bf b, v8f c) {
  // D = A(16x32 bf16) * B(32x16 bf16) + C(16x16 f32)
  return __builtin_amdgcn_wmma_f32_16x16x32_bf16(false, a, false, b, (short)0, c,
                                                 false, false);
}

DEVFN unsigned pack_bf2(float lo, float hi) {
  // -> v_cvt_pk_bf16_f32
  unsigned short a = __builtin_bit_cast(unsigned short, (__bf16)lo);
  unsigned short b = __builtin_bit_cast(unsigned short, (__bf16)hi);
  return (unsigned)a | ((unsigned)b << 16);
}

// A fragment (16x32, bf16). lds holds A row-major: lds[row][k], row stride `ld`.
// Lanes 0-15: rows M=0..15, K {0..7,16..23}; lanes 16-31: K {8..15,24..31}.
DEVFN v16bf load_a_frag(const __bf16* lds, int ld, int row0, int k0, int lane) {
  const int g = (lane >> 4) & 1, m = lane & 15;
  const __bf16* base = lds + (row0 + m) * ld + k0 + g * 8;
  v16bf a;
#pragma unroll
  for (int p = 0; p < 8; ++p) {
    const int kb = (p < 4) ? (2 * p) : (16 + 2 * (p - 4));
    a[2 * p]     = base[kb];
    a[2 * p + 1] = base[kb + 1];
  }
  return a;
}

// B fragment (32x16, bf16) where B[k][n] is stored TRANSPOSED in LDS: ldsT[n][k].
// Lanes 0-15: cols N=0..15, K {0..7,16..23}; lanes 16-31: K {8..15,24..31}.
DEVFN v16bf load_b_frag(const __bf16* ldsT, int ld, int k0, int col0, int lane) {
  const int g = (lane >> 4) & 1, n = lane & 15;
  const __bf16* base = ldsT + (col0 + n) * ld + k0 + g * 8;
  v16bf b;
#pragma unroll
  for (int p = 0; p < 8; ++p) {
    const int kb = (p < 4) ? (2 * p) : (16 + 2 * (p - 4));
    b[2 * p]     = base[kb];
    b[2 * p + 1] = base[kb + 1];
  }
  return b;
}

// Pack a C/D fragment (+bias) into one b128 transposed-LDS store:
// lane's 8 values (r=0..7) are rows (wid*16 + 8g + r) -> contiguous in dstT[d][row].
DEVFN void store_cd_transposed(__bf16* dstT, int d, int row_base, const v8f acc,
                               float bias) {
  uint4 st;
  st.x = pack_bf2(acc[0] + bias, acc[1] + bias);
  st.y = pack_bf2(acc[2] + bias, acc[3] + bias);
  st.z = pack_bf2(acc[4] + bias, acc[5] + bias);
  st.w = pack_bf2(acc[6] + bias, acc[7] + bias);
  *(uint4*)(dstT + d * 128 + row_base) = st;   // 16B-aligned
}

// ---------------------------------------------------------------------------
// Kernel 1: SLABS x 128-row slabs per block: K,V projections (WMMA),
// KV += K^T V accumulated in WMMA C-registers across slabs, then one round of
// f32 atomics.  K/V never touch HBM.
// ---------------------------------------------------------------------------
__global__ void __launch_bounds__(256)
proj_kv_kernel(const float* __restrict__ x,
               const float* __restrict__ Wk, const float* __restrict__ bk,
               const float* __restrict__ Wv, const float* __restrict__ bv,
               float* __restrict__ KVacc,   // [B][128][128] f32, pre-zeroed
               float* __restrict__ Ksum) {  // [B][128]      f32, pre-zeroed
  extern __shared__ char smem[];
  __bf16* xh  = (__bf16*)smem;          // x slab  [row][c]   32KB
  __bf16* wk  = xh + 128 * 128;         // Wk      [d][c]     32KB (== B^T)
  __bf16* wv  = wk + 128 * 128;         // Wv      [d][c]     32KB
  __bf16* kt  = wv + 128 * 128;         // K^T     [d][n]     32KB
  __bf16* vt  = kt + 128 * 128;         // V^T     [c][n]     32KB
  float*  bkl = (float*)(vt + 128 * 128);
  float*  bvl = bkl + 128;

  const int tid = threadIdx.x;
  const int b   = blockIdx.y;

  __builtin_prefetch(Wk + tid * 64, 0, 2);   // global_prefetch_b8
  __builtin_prefetch(Wv + tid * 64, 0, 2);

  // ---- weights -> bf16 LDS (once per block) ----
  const float4* wks = (const float4*)Wk;
  const float4* wvs = (const float4*)Wv;
#pragma unroll 4
  for (int i = tid; i < 4096; i += 256) {
    float4 a = wks[i]; float4 c = wvs[i];
    const int e = 4 * i;
    wk[e] = (__bf16)a.x; wk[e + 1] = (__bf16)a.y;
    wk[e + 2] = (__bf16)a.z; wk[e + 3] = (__bf16)a.w;
    wv[e] = (__bf16)c.x; wv[e + 1] = (__bf16)c.y;
    wv[e + 2] = (__bf16)c.z; wv[e + 3] = (__bf16)c.w;
  }
  if (tid < 128) { bkl[tid] = bk[tid]; bvl[tid] = bv[tid]; }

  const int wid = tid >> 5, lane = tid & 31;
  const int col = lane & 15, g = (lane >> 4) & 1;
  const int trow = wid * 16 + 8 * g;     // transposed-store row base for this lane

  v8f   kvreg[8];
#pragma unroll
  for (int ct = 0; ct < 8; ++ct) kvreg[ct] = (v8f){};
  float ksacc = 0.f;

  for (int slab = 0; slab < SLABS; ++slab) {
    // ---- x slab -> bf16 LDS ----
    const size_t row0 = (size_t)(blockIdx.x * SLABS + slab) * 128;
    const float4* xs = (const float4*)(x + ((size_t)b * Nn + row0) * Cn);
#pragma unroll 4
    for (int i = tid; i < 4096; i += 256) {
      float4 v = xs[i];
      const int e = 4 * i;
      xh[e] = (__bf16)v.x; xh[e + 1] = (__bf16)v.y;
      xh[e + 2] = (__bf16)v.z; xh[e + 3] = (__bf16)v.w;
    }
    __syncthreads();   // xh ready; prior slab's kt/vt reads finished

    // ---- K,V projections for this wave's 16-row sub-slab ----
    v16bf xa[4];
#pragma unroll
    for (int kc = 0; kc < 4; ++kc)
      xa[kc] = load_a_frag(xh, 128, wid * 16, kc * 32, lane);

#pragma unroll
    for (int dt = 0; dt < 8; ++dt) {
      v8f ak = {}; v8f av = {};
#pragma unroll
      for (int kc = 0; kc < 4; ++kc) {
        ak = wmma_bf16(xa[kc], load_b_frag(wk, 128, kc * 32, dt * 16, lane), ak);
        av = wmma_bf16(xa[kc], load_b_frag(wv, 128, kc * 32, dt * 16, lane), av);
      }
      const int d = dt * 16 + col;
      store_cd_transposed(kt, d, trow, ak, bkl[d]);
      store_cd_transposed(vt, d, trow, av, bvl[d]);
    }
    __syncthreads();   // kt/vt complete

    // ---- KV[d][c] += sum_n K[n][d] V[n][c]  (wave wid owns d-tile wid) ----
    v16bf ka[4];
#pragma unroll
    for (int kc = 0; kc < 4; ++kc)
      ka[kc] = load_a_frag(kt, 128, wid * 16, kc * 32, lane);
#pragma unroll
    for (int ct = 0; ct < 8; ++ct) {
#pragma unroll
      for (int kc = 0; kc < 4; ++kc)
        kvreg[ct] = wmma_bf16(ka[kc], load_b_frag(vt, 128, kc * 32, ct * 16, lane),
                              kvreg[ct]);
    }

    // ---- Ksum partial ----
    if (tid < 128) {
#pragma unroll 8
      for (int n = 0; n < 128; ++n) ksacc += (float)kt[tid * 128 + n];
    }
  }

  // ---- one round of global f32 atomics per block ----
  float* kvb = KVacc + (size_t)b * 128 * 128;
#pragma unroll
  for (int ct = 0; ct < 8; ++ct) {
#pragma unroll
    for (int r = 0; r < 8; ++r) {
      const int d = wid * 16 + r + 8 * g;
      atomicAdd(&kvb[d * 128 + ct * 16 + col], kvreg[ct][r]);
    }
  }
  if (tid < 128) atomicAdd(&Ksum[b * 128 + tid], ksacc);
}

// ---------------------------------------------------------------------------
// Kernel 2: KVbT[b][c][d] = bf16( KV[b][d][c] / N )
// ---------------------------------------------------------------------------
__global__ void __launch_bounds__(256)
kv_finalize_kernel(const float* __restrict__ kv, __bf16* __restrict__ kvbt) {
  const int idx = blockIdx.x * 256 + threadIdx.x;   // 0 .. 8*16384-1
  const int b = idx >> 14;
  const int r = idx & 16383;
  const int d = r >> 7;
  const int c = r & 127;
  kvbt[(b << 14) + (c << 7) + d] = (__bf16)(kv[idx] * (1.0f / 65536.0f));
}

// ---------------------------------------------------------------------------
// Kernel 3: recompute Q (WMMA), denom = |Q.Ksum|, out = Q@KV (WMMA),
// residual + fused LayerNorm, coalesced b128 stores.
// ---------------------------------------------------------------------------
__global__ void __launch_bounds__(256)
attn_out_kernel(const float* __restrict__ x,
                const float* __restrict__ Wq, const float* __restrict__ bq,
                const __bf16* __restrict__ KVbT, const float* __restrict__ Ksum,
                const float* __restrict__ lnw, const float* __restrict__ lnb,
                const float* __restrict__ res_scale, float* __restrict__ out) {
  extern __shared__ char smem[];
  float*  xf   = (float*)smem;              // x tile f32 / out staging   64KB
  __bf16* xh   = (__bf16*)(xf + 128 * 128); // x tile bf16                32KB
  __bf16* wq   = xh + 128 * 128;            // Wq [d][c]                  32KB
  __bf16* kvb  = wq + 128 * 128;            // KV^T [c][d]                32KB
  __bf16* qh   = kvb + 128 * 128;           // Q slabs [row][d]           32KB
  float*  ksl  = (float*)(qh + 128 * 128);
  float*  bql  = ksl + 128;
  float*  lnwl = bql + 128;
  float*  lnbl = lnwl + 128;

  const int tid  = threadIdx.x;
  const int b    = blockIdx.y;
  const int tile = blockIdx.x;

  __builtin_prefetch(Wq + tid * 64, 0, 2);

  const float4* xs = (const float4*)(x + ((size_t)b * Nn + (size_t)tile * 128) * Cn);
#pragma unroll 4
  for (int i = tid; i < 4096; i += 256) {
    float4 v = xs[i];
    ((float4*)xf)[i] = v;
    const int e = 4 * i;
    xh[e] = (__bf16)v.x; xh[e + 1] = (__bf16)v.y;
    xh[e + 2] = (__bf16)v.z; xh[e + 3] = (__bf16)v.w;
  }
  const float4* wqs = (const float4*)Wq;
#pragma unroll 4
  for (int i = tid; i < 4096; i += 256) {
    float4 v = wqs[i];
    const int e = 4 * i;
    wq[e] = (__bf16)v.x; wq[e + 1] = (__bf16)v.y;
    wq[e + 2] = (__bf16)v.z; wq[e + 3] = (__bf16)v.w;
  }
  const uint4* kvs = (const uint4*)(KVbT + (size_t)b * 128 * 128);
  for (int i = tid; i < 2048; i += 256) ((uint4*)kvb)[i] = kvs[i];
  if (tid < 128) {
    ksl[tid]  = Ksum[b * 128 + tid];
    bql[tid]  = bq[tid];
    lnwl[tid] = lnw[tid];
    lnbl[tid] = lnb[tid];
  }
  __syncthreads();

  const int wid = tid >> 5, lane = tid & 31;
  const int col = lane & 15, g = (lane >> 4) & 1;
  const float rs = res_scale[0];

  // ---- Q = x Wq^T + bq for this wave's 16-row slab ----
  v16bf xa[4];
#pragma unroll
  for (int kc = 0; kc < 4; ++kc)
    xa[kc] = load_a_frag(xh, 128, wid * 16, kc * 32, lane);

  v8f q[8];
#pragma unroll
  for (int dt = 0; dt < 8; ++dt) {
    v8f acc = {};
#pragma unroll
    for (int kc = 0; kc < 4; ++kc)
      acc = wmma_bf16(xa[kc], load_b_frag(wq, 128, kc * 32, dt * 16, lane), acc);
    const float bias = bql[dt * 16 + col];
#pragma unroll
    for (int r = 0; r < 8; ++r) acc[r] += bias;
    q[dt] = acc;
  }

  // ---- denom per row: |Q_row . Ksum| via half-wave xor reduction ----
  float invden[8];
#pragma unroll
  for (int r = 0; r < 8; ++r) {
    float p = 0.f;
#pragma unroll
    for (int dt = 0; dt < 8; ++dt) p += q[dt][r] * ksl[dt * 16 + col];
#pragma unroll
    for (int m = 1; m < 16; m <<= 1) p += __shfl_xor(p, m, 32);
    invden[r] = 1.0f / fmaxf(fabsf(p), 1e-6f);
  }

  // ---- Q -> bf16 LDS (row-major [row][d]) for the second GEMM's A-frags ----
  __bf16* qw = qh + wid * 16 * 128;
#pragma unroll
  for (int dt = 0; dt < 8; ++dt)
#pragma unroll
    for (int r = 0; r < 8; ++r)
      qw[(r + 8 * g) * 128 + dt * 16 + col] = (__bf16)q[dt][r];

  // ---- out = Q @ KV, /denom, *res_scale, + x ; LN stats (pass 1) ----
  v16bf qa[4];
#pragma unroll
  for (int kc = 0; kc < 4; ++kc)
    qa[kc] = load_a_frag(qw, 128, 0, kc * 32, lane);

  float s1[8], s2[8];
#pragma unroll
  for (int r = 0; r < 8; ++r) { s1[r] = 0.f; s2[r] = 0.f; }
#pragma unroll
  for (int ct = 0; ct < 8; ++ct) {
    v8f acc = {};
#pragma unroll
    for (int kc = 0; kc < 4; ++kc)
      acc = wmma_bf16(qa[kc], load_b_frag(kvb, 128, kc * 32, ct * 16, lane), acc);
    const int c = ct * 16 + col;
#pragma unroll
    for (int r = 0; r < 8; ++r) {
      const int row = wid * 16 + r + 8 * g;
      float v = acc[r] * invden[r] * rs + xf[row * 128 + c];
      xf[row * 128 + c] = v;      // each element owned by exactly one lane slot
      s1[r] += v; s2[r] += v * v;
    }
  }

  // ---- LN row statistics (biased variance, eps=1e-5) ----
  float mu[8], rstd[8];
#pragma unroll
  for (int r = 0; r < 8; ++r) {
    float a = s1[r], b2 = s2[r];
#pragma unroll
    for (int m = 1; m < 16; m <<= 1) {
      a  += __shfl_xor(a, m, 32);
      b2 += __shfl_xor(b2, m, 32);
    }
    const float mean = a * (1.0f / 128.0f);
    const float var  = b2 * (1.0f / 128.0f) - mean * mean;
    mu[r] = mean;
    rstd[r] = rsqrtf(var + 1e-5f);
  }

  // ---- pass 2: normalize in LDS ----
#pragma unroll
  for (int ct = 0; ct < 8; ++ct) {
    const int c = ct * 16 + col;
    const float w = lnwl[c], bb = lnbl[c];
#pragma unroll
    for (int r = 0; r < 8; ++r) {
      const int row = wid * 16 + r + 8 * g;
      const float v = xf[row * 128 + c];
      xf[row * 128 + c] = (v - mu[r]) * rstd[r] * w + bb;
    }
  }
  __syncthreads();

  // ---- coalesced b128 stores ----
  float4* os = (float4*)(out + ((size_t)b * Nn + (size_t)tile * 128) * Cn);
#pragma unroll 4
  for (int i = tid; i < 4096; i += 256) os[i] = ((float4*)xf)[i];
}

// ---------------------------------------------------------------------------
// Launcher
// ---------------------------------------------------------------------------
extern "C" void kernel_launch(void* const* d_in, const int* in_sizes, int n_in,
                              void* d_out, int out_size, void* d_ws, size_t ws_size,
                              hipStream_t stream) {
  (void)in_sizes; (void)n_in; (void)out_size; (void)ws_size;
  const float* x   = (const float*)d_in[0];
  const float* Wq  = (const float*)d_in[1];
  const float* bq  = (const float*)d_in[2];
  const float* Wk  = (const float*)d_in[3];
  const float* bk  = (const float*)d_in[4];
  const float* Wv  = (const float*)d_in[5];
  const float* bv  = (const float*)d_in[6];
  const float* lnw = (const float*)d_in[7];
  const float* lnb = (const float*)d_in[8];
  const float* rs  = (const float*)d_in[9];

  float*  KVacc = (float*)d_ws;                       // 8*128*128 f32
  float*  Ksum  = KVacc + (size_t)Bn * 128 * 128;     // 8*128 f32
  __bf16* KVbT  = (__bf16*)(Ksum + (size_t)Bn * 128); // 8*128*128 bf16

  // zero the accumulators every call (capture-safe memset node)
  hipMemsetAsync(d_ws, 0,
                 ((size_t)Bn * 128 * 128 + (size_t)Bn * 128) * sizeof(float),
                 stream);

  const size_t ldsA = 5u * 128 * 128 * sizeof(__bf16) + 2u * 128 * sizeof(float);
  const size_t ldsC = 128u * 128 * sizeof(float) + 4u * 128 * 128 * sizeof(__bf16)
                    + 4u * 128 * sizeof(float);
  (void)hipFuncSetAttribute((const void*)proj_kv_kernel,
                            hipFuncAttributeMaxDynamicSharedMemorySize, (int)ldsA);
  (void)hipFuncSetAttribute((const void*)attn_out_kernel,
                            hipFuncAttributeMaxDynamicSharedMemorySize, (int)ldsC);

  proj_kv_kernel<<<dim3(Nn / (128 * SLABS), Bn), 256, ldsA, stream>>>(
      x, Wk, bk, Wv, bv, KVacc, Ksum);
  kv_finalize_kernel<<<dim3((Bn * 128 * 128) / 256), 256, 0, stream>>>(KVacc, KVbT);
  attn_out_kernel<<<dim3(Nn / 128, Bn), 256, ldsC, stream>>>(
      x, Wq, bq, KVbT, Ksum, lnw, lnb, rs, (float*)d_out);
}
